// DeformableAlignment_79259326480632
// MI455X (gfx1250) — matrix operand order repaired and checked by reading, not compiled, not run on hardware
//
#include <hip/hip_runtime.h>
#include <hip/hip_bf16.h>
#include <math.h>

typedef __attribute__((ext_vector_type(16))) _Float16 v16h;
typedef __attribute__((ext_vector_type(8)))  float    v8f;
typedef __attribute__((ext_vector_type(4)))  float    v4f;

#define HH 64
#define WW 64
#define CC 128
#define BB 8
#define HW 4096          // 64*64
#define KSTEPS 72        // K=2304 (= 9 taps * 256 ch) in steps of 32

// ---------------------------------------------------------------------------
// Phase 1a: partial column sums of similarity_map [B, 4096, 4096] over axis 1.
// s[b, j] = sum_i sim[b, i, j].  Deterministic two-phase (no float atomics).
// Non-temporal loads: 512 MB read-once stream must not sweep the 192 MB L2
// (x and the conv outputs need to stay resident for the later phases).
// Block: 1024 threads (each owns one float4 column group => full 4096 cols).
// Grid: 8 batches * 32 row-chunks of 128 rows.
// ---------------------------------------------------------------------------
__global__ __launch_bounds__(1024)
void sim_partial_kernel(const float* __restrict__ sim, float* __restrict__ partial) {
    int b     = blockIdx.x & 7;
    int chunk = blockIdx.x >> 3;           // 0..31
    int g     = threadIdx.x;               // 0..1023 float4 column group
    const v4f* row = (const v4f*)(sim + ((size_t)b * HW + (size_t)chunk * 128) * HW);
    v4f acc = {0.f, 0.f, 0.f, 0.f};
    for (int r = 0; r < 128; ++r) {
        v4f v = __builtin_nontemporal_load(row + (size_t)r * 1024 + g);
        acc += v;
    }
    __builtin_nontemporal_store(acc, (v4f*)(partial + ((size_t)chunk * 8 + b) * HW) + g);
}

// Phase 1b: s[i] = sum over 32 chunks. i = b*4096 + j, 32768 threads.
__global__ __launch_bounds__(256)
void sim_reduce_kernel(const float* __restrict__ partial, float* __restrict__ s) {
    int i = blockIdx.x * blockDim.x + threadIdx.x;   // < 32768
    float acc = 0.f;
    #pragma unroll 8
    for (int c = 0; c < 32; ++c)
        acc += __builtin_nontemporal_load(partial + (size_t)c * 32768 + i);
    s[i] = acc;
}

// ---------------------------------------------------------------------------
// Phase 2a: pack conv weights (OIHW f32) into per-lane WMMA B fragments (f16).
// Layout: wpack[((ntile*72 + t)*32 + lane)*16 + j]
//   B-fragment (16-bit 32x16, wave32): lane l -> N = l%16,
//   K = j + 16*(l/16)   (VGPR r = j/2, low/high half = j%2).
// Global k = 32*t + K ; tap = k/256, cin2 = k%256.
// n = l%16 + 16*ntile: n<18 -> w_off, n<27 -> w_mod, else 0 (pad).
// ---------------------------------------------------------------------------
__global__ __launch_bounds__(256)
void pack_weights_kernel(const float* __restrict__ w_off, const float* __restrict__ w_mod,
                         _Float16* __restrict__ wpack) {
    int id = blockIdx.x * blockDim.x + threadIdx.x;  // < 2*72*32*16 = 73728
    int j   = id & 15;
    int tmp = id >> 4;
    int l   = tmp & 31; tmp >>= 5;
    int t   = tmp % KSTEPS;
    int ntile = tmp / KSTEPS;
    int half  = l >> 4;
    int n     = (l & 15) + ntile * 16;
    int kk    = j + 16 * half;               // 0..31
    int tap   = t >> 3;                      // 32*t + kk spans one tap
    int cin2  = ((t & 7) << 5) + kk;         // 0..255
    float v = 0.f;
    if (n < 18)       v = w_off[((size_t)n * 256 + cin2) * 9 + tap];
    else if (n < 27)  v = w_mod[((size_t)(n - 18) * 256 + cin2) * 9 + tap];
    wpack[id] = (_Float16)v;
}

// ---------------------------------------------------------------------------
// Phase 2b: implicit-GEMM 3x3 conv via v_wmma_f32_16x16x32_f16.
// One wave per output tile: (b, h, 16 consecutive w).  2048 waves.
// feat channel c<128 = x[b,c,*], c>=128 = x[b,c-128,*] * s[b,*] (on the fly).
// Bias folded into the accumulator initialization.
// Output: convout[b][n][h][w], n = 0..17 offsets, 18..26 pre-sigmoid mod.
// ---------------------------------------------------------------------------
__global__ __launch_bounds__(256)
void conv_wmma_kernel(const float* __restrict__ x, const float* __restrict__ s,
                      const _Float16* __restrict__ wpack,
                      const float* __restrict__ b_off, const float* __restrict__ b_mod,
                      float* __restrict__ convout) {
    int wave = (blockIdx.x * blockDim.x + threadIdx.x) >> 5;  // 0..2047
    int lane = threadIdx.x & 31;
    int b    = wave >> 8;
    int rem  = wave & 255;
    int h    = rem >> 2;
    int w0   = (rem & 3) << 4;
    int half = lane >> 4;
    int m    = lane & 15;       // A-fragment row M for this lane
    int wpix = w0 + m;

    int n0 = lane & 15;         // C/D fragment column N for this lane
    int n1 = n0 + 16;
    float bias0 = (n0 < 18) ? b_off[n0] : b_mod[n0 - 18];
    float bias1 = (n1 < 18) ? b_off[n1] : (n1 < 27 ? b_mod[n1 - 18] : 0.f);
    v8f acc0 = {bias0, bias0, bias0, bias0, bias0, bias0, bias0, bias0};
    v8f acc1 = {bias1, bias1, bias1, bias1, bias1, bias1, bias1, bias1};

    const float* xb = x + (size_t)b * CC * HW;
    const float* sb = s + (size_t)b * HW;

    for (int t = 0; t < KSTEPS; ++t) {
        int tap     = t >> 3;
        int cinBase = (t & 7) << 5;          // 0..224
        int dh = tap / 3 - 1, dw = tap % 3 - 1;
        int hh = h + dh, ww = wpix + dw;
        bool inb = ((unsigned)hh < 64u) && ((unsigned)ww < 64u);
        int sp = hh * 64 + ww;
        bool second = cinBase >= 128;        // weighted_x half of feat
        float sv = 1.f;
        if (second && inb) sv = sb[sp];
        int cbase = second ? (cinBase - 128) : cinBase;

        // A fragment, 16-bit 16x32 layout (matches packed B):
        //   lane<16:  j=0..15 -> K=j       (K 0..15)
        //   lane>=16: add 16               (K 16..31)
        v16h a;
        #pragma unroll
        for (int j = 0; j < 16; ++j) {
            int kk = j + 16 * half;
            int c  = cbase + kk;
            float v = inb ? xb[(size_t)c * HW + sp] * sv : 0.f;
            a[j] = (_Float16)v;
        }

        const v16h bw0 = *(const v16h*)(wpack + ((size_t)(0 * KSTEPS + t) * 32 + lane) * 16);
        const v16h bw1 = *(const v16h*)(wpack + ((size_t)(1 * KSTEPS + t) * 32 + lane) * 16);

        acc0 = __builtin_amdgcn_wmma_f32_16x16x32_f16(false, a, false, bw0,
                                                      (short)0, acc0, false, false);
        acc1 = __builtin_amdgcn_wmma_f32_16x16x32_f16(false, a, false, bw1,
                                                      (short)0, acc1, false, false);
    }

    // C/D layout: lane l, VGPR r -> (M = r + 8*(l>=16), N = l%16)
    float* cb = convout + (size_t)b * 27 * HW;
    #pragma unroll
    for (int r = 0; r < 8; ++r) {
        int M   = r + 8 * half;
        int pix = h * 64 + w0 + M;
        cb[(size_t)n0 * HW + pix] = acc0[r];
        if (n1 < 27) cb[(size_t)n1 * HW + pix] = acc1[r];
    }
}

// ---------------------------------------------------------------------------
// Phase 3a: bilinear coefficients per (b, tap, pixel).  294912 threads.
// coef4 = {c00,c10,c01,c11} * sigmoid(mod) * in-bounds mask (AoS float4 so
// the sampling kernel fetches all four with a single b128); corners packed
// into one int.
// ---------------------------------------------------------------------------
__global__ __launch_bounds__(256)
void coef_kernel(const float* __restrict__ convout,
                 v4f* __restrict__ coef4, int* __restrict__ idx) {
    int id = blockIdx.x * blockDim.x + threadIdx.x;  // < 8*9*4096
    int p = id & 4095;
    int k = (id >> 12) % 9;
    int b = id / (9 * 4096);
    const float* cb = convout + (size_t)b * 27 * HW;
    float offh = cb[(size_t)(2 * k) * HW + p];
    float offw = cb[(size_t)(2 * k + 1) * HW + p];
    float modr = cb[(size_t)(18 + k) * HW + p];
    int h = p >> 6, w = p & 63;
    float oh = (float)h + offh;
    float ow = (float)w + offw;
    float mask = (oh >= 0.f && oh <= 63.f && ow >= 0.f && ow <= 63.f) ? 1.f : 0.f;
    float mod  = mask / (1.f + expf(-modr));
    int h0 = (int)fminf(fmaxf(floorf(oh), 0.f), 63.f);
    int w0 = (int)fminf(fmaxf(floorf(ow), 0.f), 63.f);
    int h1 = min(h0 + 1, 63);
    int w1 = min(w0 + 1, 63);
    float lh = oh - (float)h0;
    float lw = ow - (float)w0;
    v4f c4;
    c4.x = (1.f - lh) * (1.f - lw) * mod;
    c4.y = lh * (1.f - lw) * mod;
    c4.z = (1.f - lh) * lw * mod;
    c4.w = lh * lw * mod;
    coef4[id] = c4;
    idx[id] = h0 | (w0 << 8) | (h1 << 16) | (w1 << 24);
}

// ---------------------------------------------------------------------------
// Phase 3b: gather + accumulate.  Thread per (b, c, pixel) = 4.19M threads.
// Per tap: 1 b128 coef load + 1 b32 idx load + 4 gathers from L2-resident x.
// ---------------------------------------------------------------------------
__global__ __launch_bounds__(256)
void sample_kernel(const float* __restrict__ x,
                   const v4f* __restrict__ coef4, const int* __restrict__ idx,
                   float* __restrict__ out) {
    int id = blockIdx.x * blockDim.x + threadIdx.x;  // < 8*128*4096
    int p = id & 4095;
    int c = (id >> 12) & 127;
    int b = id >> 19;
    const float* xb = x + ((size_t)b * CC + c) * HW;
    float acc = 0.f;
    #pragma unroll
    for (int k = 0; k < 9; ++k) {
        size_t q = ((size_t)b * 9 + k) * HW + p;
        int ip = idx[q];
        v4f cc = coef4[q];
        int h0 = ip & 255, w0 = (ip >> 8) & 255, h1 = (ip >> 16) & 255, w1 = (ip >> 24) & 255;
        float v00 = xb[h0 * 64 + w0];
        float v10 = xb[h1 * 64 + w0];
        float v01 = xb[h0 * 64 + w1];
        float v11 = xb[h1 * 64 + w1];
        acc += cc.x * v00 + cc.y * v10 + cc.z * v01 + cc.w * v11;
    }
    out[((size_t)b * CC + c) * HW + p] = acc;
}

// ---------------------------------------------------------------------------
extern "C" void kernel_launch(void* const* d_in, const int* in_sizes, int n_in,
                              void* d_out, int out_size, void* d_ws, size_t ws_size,
                              hipStream_t stream) {
    const float* x     = (const float*)d_in[0];   // [8,128,64,64]
    const float* sim   = (const float*)d_in[1];   // [8,4096,4096]
    const float* w_off = (const float*)d_in[2];   // [18,256,3,3]
    const float* b_off = (const float*)d_in[3];   // [18]
    const float* w_mod = (const float*)d_in[4];   // [9,256,3,3]
    const float* b_mod = (const float*)d_in[5];   // [9]
    float* out = (float*)d_out;

    char* ws = (char*)d_ws;
    float*    partial = (float*)(ws + 0);            // 32*8*4096 f32  = 4,194,304 B
    float*    sbuf    = (float*)(ws + 4194304);      // 8*4096 f32     =   131,072 B
    _Float16* wpack   = (_Float16*)(ws + 4325376);   // 73728 f16      =   147,456 B
    float*    convout = (float*)(ws + 4472832);      // 8*27*4096 f32  = 3,538,944 B
    v4f*      coef4   = (v4f*)(ws + 8011776);        // 294912 v4f     = 4,718,592 B
    int*      idxbuf  = (int*)(ws + 12730368);       // 294912 i32     = 1,179,648 B
                                                     // total          = 13,910,016 B

    // 1) similarity column sum (512 MB NT stream -> dominant, HBM-bound)
    sim_partial_kernel<<<256, 1024, 0, stream>>>(sim, partial);
    sim_reduce_kernel<<<128, 256, 0, stream>>>(partial, sbuf);

    // 2) conv via WMMA
    pack_weights_kernel<<<288, 256, 0, stream>>>(w_off, w_mod, wpack);
    conv_wmma_kernel<<<256, 256, 0, stream>>>(x, sbuf, wpack, b_off, b_mod, convout);

    // 3) deformable bilinear sampling
    coef_kernel<<<1152, 256, 0, stream>>>(convout, coef4, idxbuf);
    sample_kernel<<<16384, 256, 0, stream>>>(x, coef4, idxbuf, out);
}
